// SlotAttention_41558103556448
// MI455X (gfx1250) — matrix operand order repaired and checked by reading, not compiled, not run on hardware
//
#include <hip/hip_runtime.h>

#define BB 128
#define NN 4096
#define DD 64
#define SS 7
#define HH 128

typedef __bf16 bf16;
typedef __attribute__((ext_vector_type(16))) __bf16 v16bf;
typedef __attribute__((ext_vector_type(8)))  __bf16 v8bf;
typedef __attribute__((ext_vector_type(8)))  float  v8f;

// ---- gfx1250 async memory->LDS (ASYNCcnt-tracked), per CDNA5 ISA ----
static __device__ inline void async_ld_b128(unsigned ldsOff, unsigned long long gaddr) {
  asm volatile("global_load_async_to_lds_b128 %0, %1, off" :: "v"(ldsOff), "v"(gaddr) : "memory");
}
static __device__ inline void async_ld_b64(unsigned ldsOff, unsigned long long gaddr) {
  asm volatile("global_load_async_to_lds_b64 %0, %1, off" :: "v"(ldsOff), "v"(gaddr) : "memory");
}
static __device__ inline void wait_async0() {
  asm volatile("s_wait_asynccnt 0" ::: "memory");
}
static __device__ inline unsigned lds_off(const void* p) {
  // LDS aperture: generic addr low 32 bits == LDS byte offset (ISA 10.2)
  return (unsigned)(unsigned long long)p;
}

// ---- WMMA operand gathers (CDNA5 ISA 7.12.2 VGPR layouts) ----
// A 16-bit 16x32: lane holds row m=L&15; its 16 values are two contiguous runs of 8:
//   K = half*8+k0 .. +7  and  K = 16+half*8+k0 .. +7   -> two ds_load_b128
static __device__ inline v16bf gather_a(const bf16* base, int stride, int m, int half, int k0) {
  v8bf lo = *(const v8bf*)(base + m * stride + half * 8 + k0);
  v8bf hi = *(const v8bf*)(base + m * stride + 16 + half * 8 + k0);
  return __builtin_shufflevector(lo, hi, 0, 1, 2, 3, 4, 5, 6, 7, 8, 9, 10, 11, 12, 13, 14, 15);
}
// B 16-bit 32x16: lane holds col n=L&15, 16 contiguous values from rowT (operand staged transposed)
static __device__ inline v16bf load_brow(const bf16* rowT) {
  v8bf lo = *(const v8bf*)(rowT);
  v8bf hi = *(const v8bf*)(rowT + 8);
  return __builtin_shufflevector(lo, hi, 0, 1, 2, 3, 4, 5, 6, 7, 8, 9, 10, 11, 12, 13, 14, 15);
}

// ---- Kernel 0: slot init: mu + exp(log_sigma)*noise ----
__global__ void k_init(const float* __restrict__ nbg, const float* __restrict__ nfg,
                       const float* __restrict__ bgmu, const float* __restrict__ bgls,
                       const float* __restrict__ smu, const float* __restrict__ sls,
                       float* __restrict__ slots) {
  int b = blockIdx.x, t = threadIdx.x;     // 448 threads = 7*64
  int s = t >> 6, d = t & 63;
  float v;
  if (s == 0) v = bgmu[d] + expf(bgls[d]) * nbg[(size_t)b * 64 + d];
  else        v = smu[d]  + expf(sls[d])  * nfg[((size_t)b * 6 + (s - 1)) * 64 + d];
  slots[(size_t)b * (SS * 64) + t] = v;
}

// ---- Kernel 1: fused LN(x) + k/v projection (WMMA), outputs bf16 ----
__global__ void __launch_bounds__(256) k_ln_proj(
    const float* __restrict__ x, const float* __restrict__ g, const float* __restrict__ bt,
    const float* __restrict__ Wk, const float* __restrict__ Wv,
    bf16* __restrict__ kbuf, bf16* __restrict__ vbuf) {
  __shared__ __align__(16) bf16 sWkT[64 * 64];   // W layout [e][d]: row e = B col e
  __shared__ __align__(16) bf16 sWvT[64 * 64];
  __shared__ __align__(16) bf16 sX[8][16 * 64];
  int tid = threadIdx.x;
  int b = blockIdx.x >> 5, nblk = blockIdx.x & 31;
  int w = tid >> 5, lane = tid & 31;
  int n0 = nblk * 128 + w * 16;
  for (int i = tid; i < 4096; i += 256) {
    int e = i >> 6, d = i & 63;                  // W row-major [e][d]
    sWkT[e * 64 + d] = (bf16)Wk[i];
    sWvT[e * 64 + d] = (bf16)Wv[i];
  }
  if (lane < 16) {
    const float* xr = x + ((size_t)(b * NN + n0 + lane)) * 64;
    float mu = 0.f;
#pragma unroll
    for (int d = 0; d < 64; ++d) mu += xr[d];
    mu *= (1.f / 64.f);
    float var = 0.f;
#pragma unroll
    for (int d = 0; d < 64; ++d) { float t = xr[d] - mu; var += t * t; }
    var *= (1.f / 64.f);
    float rs = rsqrtf(var + 1e-5f);
#pragma unroll
    for (int d = 0; d < 64; ++d)
      sX[w][lane * 64 + d] = (bf16)((xr[d] - mu) * rs * g[d] + bt[d]);
  }
  __syncthreads();
  int half = lane >> 4, col = lane & 15, m16 = lane & 15;
  const bf16* xs = sX[w];
#pragma unroll
  for (int t = 0; t < 2; ++t) {
    const bf16* wmat = t ? sWvT : sWkT;
    bf16* outb = t ? vbuf : kbuf;
#pragma unroll
    for (int nt = 0; nt < 4; ++nt) {
      v8f c = {};
#pragma unroll
      for (int kt = 0; kt < 2; ++kt) {
        v16bf a  = gather_a(xs, 64, m16, half, kt * 32);
        v16bf bm = gather_a(wmat + nt * 16 * 64, 64, col, half, kt * 32);
        c = __builtin_amdgcn_wmma_f32_16x16x32_bf16(false, a, false, bm, (short)0, c, false, false);
      }
#pragma unroll
      for (int r = 0; r < 8; ++r) {
        int m = r + 8 * half;
        outb[((size_t)(b * NN + n0 + m)) * 64 + nt * 16 + col] = (bf16)c[r];
      }
    }
  }
}

// ---- Kernel 2: s = LN(slots), q = s @ Wq^T (WMMA); store q bf16 row-major [s][d] ----
__global__ void __launch_bounds__(128) k_slots_q(
    const float* __restrict__ slots, const float* __restrict__ g, const float* __restrict__ bt,
    const float* __restrict__ Wq, bf16* __restrict__ qbufT) {
  __shared__ __align__(16) bf16 sWqT[64 * 64];
  __shared__ __align__(16) bf16 sS[16 * 64];
  int b = blockIdx.x, tid = threadIdx.x;
  for (int i = tid; i < 4096; i += 128) {
    int e = i >> 6, d = i & 63;
    sWqT[e * 64 + d] = (bf16)Wq[i];
  }
  if (tid < 16) {
    if (tid < SS) {
      const float* sr = slots + ((size_t)b * SS + tid) * 64;
      float mu = 0.f;
#pragma unroll
      for (int d = 0; d < 64; ++d) mu += sr[d];
      mu *= (1.f / 64.f);
      float var = 0.f;
#pragma unroll
      for (int d = 0; d < 64; ++d) { float t = sr[d] - mu; var += t * t; }
      var *= (1.f / 64.f);
      float rs = rsqrtf(var + 1e-5f);
#pragma unroll
      for (int d = 0; d < 64; ++d)
        sS[tid * 64 + d] = (bf16)((sr[d] - mu) * rs * g[d] + bt[d]);
    } else {
#pragma unroll
      for (int d = 0; d < 64; ++d) sS[tid * 64 + d] = (bf16)0.f;
    }
  }
  __syncthreads();
  if (tid < 32) {
    int lane = tid, half = lane >> 4, m16 = lane & 15, col = lane & 15;
#pragma unroll
    for (int nt = 0; nt < 4; ++nt) {
      v8f c = {};
#pragma unroll
      for (int kt = 0; kt < 2; ++kt) {
        v16bf a  = gather_a(sS, 64, m16, half, kt * 32);
        v16bf bm = gather_a(sWqT + nt * 16 * 64, 64, col, half, kt * 32);
        c = __builtin_amdgcn_wmma_f32_16x16x32_bf16(false, a, false, bm, (short)0, c, false, false);
      }
#pragma unroll
      for (int r = 0; r < 8; ++r)  // q[s][d], d = nt*16+col, s = r+8*half
        qbufT[(size_t)b * 1024 + (r + 8 * half) * 64 + (nt * 16 + col)] = (bf16)c[r];
    }
  }
}

// ---- Kernel 3: logits = scale * k q^T (WMMA), softmax over slots, +eps, attn bf16 ----
__global__ void __launch_bounds__(256) k_attn(
    const bf16* __restrict__ kbuf, const bf16* __restrict__ qbufT, const bf16* __restrict__ vbuf,
    bf16* __restrict__ attnb) {
  __shared__ __align__(16) bf16 sQT[64 * 16];   // [s][d], row-major q as stored in global
  __shared__ __align__(16) bf16 sK[8][16 * 64];
  __shared__ __align__(16) float sL[8][16 * 16];
  int tid = threadIdx.x, b = blockIdx.x >> 5, nblk = blockIdx.x & 31;
  int w = tid >> 5, lane = tid & 31;
  int n0 = nblk * 128 + w * 16;
  // async copy qT (2KB, whole block) and this wave's k tile (2KB) into LDS
  {
    unsigned qlds = lds_off(sQT);
    unsigned long long qg = (unsigned long long)(const void*)(qbufT + (size_t)b * 1024);
    async_ld_b64(qlds + tid * 8u, qg + tid * 8u);
    unsigned klds = lds_off(sK[w]);
    unsigned long long kg = (unsigned long long)(const void*)(kbuf + ((size_t)(b * NN + n0)) * 64);
#pragma unroll
    for (int cidx = 0; cidx < 4; ++cidx) {
      unsigned off = (unsigned)(cidx * 32 + lane) * 16u;
      async_ld_b128(klds + off, kg + off);
    }
    // warm caches for the v tile the updates kernel will stream next
    __builtin_prefetch(vbuf + ((size_t)(b * NN + n0)) * 64 + (size_t)lane * 64, 0, 1);
  }
  wait_async0();
  __syncthreads();
  int half = lane >> 4, col = lane & 15, m16 = lane & 15;
  v8f c = {};
#pragma unroll
  for (int kt = 0; kt < 2; ++kt) {
    v16bf a  = gather_a(sK[w], 64, m16, half, kt * 32);
    v16bf bm = gather_a(sQT, 64, col, half, kt * 32);   // B[k=d][s] = q[s][d]
    c = __builtin_amdgcn_wmma_f32_16x16x32_bf16(false, a, false, bm, (short)0, c, false, false);
  }
#pragma unroll
  for (int r = 0; r < 8; ++r) sL[w][(r + 8 * half) * 16 + col] = c[r] * 0.125f; // D^-0.5
  __syncthreads();
  if (lane < 16) {
    float* row = &sL[w][lane * 16];
    float mx = row[0];
#pragma unroll
    for (int s = 1; s < SS; ++s) mx = fmaxf(mx, row[s]);
    float e[SS], sum = 0.f;
#pragma unroll
    for (int s = 0; s < SS; ++s) { e[s] = expf(row[s] - mx); sum += e[s]; }
    float inv = 1.f / sum;
    bf16* arow = attnb + ((size_t)(b * NN + n0 + lane)) * 16;
#pragma unroll
    for (int s = 0; s < SS; ++s) arow[s] = (bf16)(e[s] * inv + 1e-8f);
#pragma unroll
    for (int s = SS; s < 16; ++s) arow[s] = (bf16)0.f;
  }
}

// ---- Kernel 4: updates = (attn / colsum_n)^T v (WMMA over K=N), per-batch block ----
__global__ void __launch_bounds__(128) k_updates(
    const bf16* __restrict__ attnb, const bf16* __restrict__ vbuf,
    float* __restrict__ ubuf) {
  __shared__ __align__(16) bf16 sAT[16 * 32];   // transposed attn tile: [s][n]
  __shared__ __align__(16) bf16 sV[32 * 64];
  __shared__ __align__(16) float sCS[16];
  int b = blockIdx.x, tid = threadIdx.x;
  int w = tid >> 5, lane = tid & 31;
  int half = lane >> 4, col = lane & 15, m16 = lane & 15;
  v8f c = {};
  float csAcc = 0.f;   // wave 0, lane<16: deterministic column sum over N
  unsigned vlds = lds_off(sV);
  for (int ks = 0; ks < 128; ++ks) {
    int n0 = ks * 32;
    {
      // async: v tile 4KB (2 x b128 per thread)
      unsigned long long vg = (unsigned long long)(const void*)(vbuf + ((size_t)(b * NN + n0)) * 64);
#pragma unroll
      for (int cidx = 0; cidx < 2; ++cidx) {
        unsigned off = (unsigned)(cidx * 128 + tid) * 16u;
        async_ld_b128(vlds + off, vg + off);
      }
      // attn tile: transpose while staging (4 elements per thread)
      const bf16* ab = attnb + ((size_t)(b * NN + n0)) * 16;
      for (int i = tid; i < 512; i += 128) {
        int n = i >> 4, s = i & 15;
        sAT[s * 32 + n] = ab[i];
      }
    }
    wait_async0();
    __syncthreads();
    v16bf a  = gather_a(sAT, 32, m16, half, 0);       // A[m=s][k=n]
    v16bf bm;                                         // B[k=n][d], d-tile = wave
#pragma unroll
    for (int s = 0; s < 16; ++s) bm[s] = sV[((s & 7) + 8 * half + 16 * (s >> 3)) * 64 + w * 16 + col];
    c = __builtin_amdgcn_wmma_f32_16x16x32_bf16(false, a, false, bm, (short)0, c, false, false);
    if (w == 0 && lane < 16) {
      float s = 0.f;
#pragma unroll
      for (int k = 0; k < 32; ++k) s += (float)sAT[lane * 32 + k];
      csAcc += s;
    }
    __syncthreads();
  }
  if (w == 0 && lane < 16) sCS[lane] = fmaxf(csAcc, 1e-20f);
  __syncthreads();
#pragma unroll
  for (int r = 0; r < 8; ++r) {
    int s = r + 8 * half;
    ubuf[(size_t)b * 1024 + s * 64 + w * 16 + col] = c[r] / sCS[s];
  }
}

// ---- Kernel 5: GRU cell + LN + MLP residual per (b,s) row ----
__global__ void __launch_bounds__(192) k_gru_mlp(
    const float* __restrict__ ubuf, float* __restrict__ slots,
    const float* __restrict__ Wih, const float* __restrict__ Whh,
    const float* __restrict__ bih, const float* __restrict__ bhh,
    const float* __restrict__ lng, const float* __restrict__ lnb,
    const float* __restrict__ w1, const float* __restrict__ b1,
    const float* __restrict__ w2, const float* __restrict__ b2,
    float* __restrict__ outp, int write_out) {
  __shared__ float sU[64], sH[64], sGi[192], sGh[192], sHn[64], sA1[128];
  int row = blockIdx.x, t = threadIdx.x;
  int b = row / SS, s = row % SS;
  if (t < 64) {
    sU[t] = ubuf[(size_t)b * 1024 + s * 64 + t];
    sH[t] = slots[(size_t)row * 64 + t];
  }
  __syncthreads();
  {
    float gi = bih[t], gh = bhh[t];
    const float* wi = Wih + (size_t)t * 64;
    const float* wh = Whh + (size_t)t * 64;
#pragma unroll 4
    for (int d = 0; d < 64; ++d) { gi += sU[d] * wi[d]; gh += sH[d] * wh[d]; }
    sGi[t] = gi; sGh[t] = gh;
  }
  __syncthreads();
  if (t < 64) {
    float r = 1.f / (1.f + expf(-(sGi[t] + sGh[t])));
    float z = 1.f / (1.f + expf(-(sGi[64 + t] + sGh[64 + t])));
    float n = tanhf(sGi[128 + t] + r * sGh[128 + t]);
    sHn[t] = (1.f - z) * n + z * sH[t];
  }
  __syncthreads();
  float mu = 0.f;
#pragma unroll 4
  for (int d = 0; d < 64; ++d) mu += sHn[d];
  mu *= (1.f / 64.f);
  float var = 0.f;
#pragma unroll 4
  for (int d = 0; d < 64; ++d) { float x = sHn[d] - mu; var += x * x; }
  var *= (1.f / 64.f);
  float rs = rsqrtf(var + 1e-5f);
  if (t < 128) {
    float acc = b1[t];
    const float* wr = w1 + (size_t)t * 64;
#pragma unroll 4
    for (int d = 0; d < 64; ++d) acc += ((sHn[d] - mu) * rs * lng[d] + lnb[d]) * wr[d];
    sA1[t] = fmaxf(acc, 0.f);
  }
  __syncthreads();
  if (t < 64) {
    float o = b2[t];
    const float* wr = w2 + (size_t)t * 128;
#pragma unroll 4
    for (int h = 0; h < 128; ++h) o += sA1[h] * wr[h];
    float ns = sHn[t] + o;
    slots[(size_t)row * 64 + t] = ns;
    if (write_out) outp[(size_t)row * 64 + t] = ns;
  }
}

extern "C" void kernel_launch(void* const* d_in, const int* in_sizes, int n_in,
                              void* d_out, int out_size, void* d_ws, size_t ws_size,
                              hipStream_t stream) {
  const float* inputs    = (const float*)d_in[0];
  const float* noise_bg  = (const float*)d_in[1];
  const float* noise_fg  = (const float*)d_in[2];
  const float* ln_in_g   = (const float*)d_in[3];
  const float* ln_in_b   = (const float*)d_in[4];
  const float* ln_slot_g = (const float*)d_in[5];
  const float* ln_slot_b = (const float*)d_in[6];
  const float* ln_mlp_g  = (const float*)d_in[7];
  const float* ln_mlp_b  = (const float*)d_in[8];
  const float* Wq        = (const float*)d_in[9];
  const float* Wk        = (const float*)d_in[10];
  const float* Wv        = (const float*)d_in[11];
  const float* W_ih      = (const float*)d_in[12];
  const float* W_hh      = (const float*)d_in[13];
  const float* b_ih      = (const float*)d_in[14];
  const float* b_hh      = (const float*)d_in[15];
  const float* mlp_w1    = (const float*)d_in[16];
  const float* mlp_b1    = (const float*)d_in[17];
  const float* mlp_w2    = (const float*)d_in[18];
  const float* mlp_b2    = (const float*)d_in[19];
  const float* bgmu      = (const float*)d_in[20];
  const float* bgls      = (const float*)d_in[21];
  const float* smu       = (const float*)d_in[22];
  const float* sls       = (const float*)d_in[23];
  float* out = (float*)d_out;

  char* ws = (char*)d_ws;
  size_t off = 0;
  bf16* kbuf  = (bf16*)(ws + off); off += (size_t)BB * NN * DD * 2;   // 64 MB
  bf16* vbuf  = (bf16*)(ws + off); off += (size_t)BB * NN * DD * 2;   // 64 MB
  bf16* attnb = (bf16*)(ws + off); off += (size_t)BB * NN * 16 * 2;   // 16 MB
  bf16* qbufT = (bf16*)(ws + off); off += (size_t)BB * 16 * 64 * 2;
  float* slots = (float*)(ws + off); off += (size_t)BB * SS * 64 * 4;
  float* ubuf  = (float*)(ws + off); off += (size_t)BB * 16 * 64 * 4;
  (void)in_sizes; (void)n_in; (void)out_size; (void)ws_size;

  k_init<<<BB, SS * 64, 0, stream>>>(noise_bg, noise_fg, bgmu, bgls, smu, sls, slots);
  k_ln_proj<<<BB * NN / 128, 256, 0, stream>>>(inputs, ln_in_g, ln_in_b, Wk, Wv, kbuf, vbuf);
  for (int it = 0; it < 3; ++it) {
    k_slots_q<<<BB, 128, 0, stream>>>(slots, ln_slot_g, ln_slot_b, Wq, qbufT);
    k_attn<<<BB * NN / 128, 256, 0, stream>>>(kbuf, qbufT, vbuf, attnb);
    k_updates<<<BB, 128, 0, stream>>>(attnb, vbuf, ubuf);
    k_gru_mlp<<<BB * SS, 192, 0, stream>>>(ubuf, slots, W_ih, W_hh, b_ih, b_hh,
        ln_mlp_g, ln_mlp_b, mlp_w1, mlp_b1, mlp_w2, mlp_b2, out, it == 2);
  }
}
